// MaxAvgPointMVSNet_82721070120981
// MI455X (gfx1250) — matrix operand order repaired and checked by reading, not compiled, not run on hardware
//
#include <hip/hip_runtime.h>

typedef __attribute__((ext_vector_type(16))) _Float16 v16h;
typedef __attribute__((ext_vector_type(8)))  float    v8f;

namespace {
constexpr int NP = 8192;   // points
constexpr int NB = 2;      // batch
constexpr int NV = 3;      // views
constexpr int NC = 112;    // feature channels
constexpr int HF = 128;
constexpr int WF = 160;
constexpr int KK = 16;     // neighbors
}

// ---- WMMA fragment index maps (contiguous k order used here for A and B, 16-bit operands) ----
// A matrix 16x32: lane<16 holds K=0..15, lane>=16 holds K=16..31 (contiguous)
__device__ __forceinline__ int kofA(int h, int hi) {
  return h + (hi ? 16 : 0);
}
// B matrix 32x16: lane<16 holds K=0..15, lane>=16 holds K=16..31 (contiguous)
__device__ __forceinline__ int kofB(int h, int hi) {
  return h + (hi ? 16 : 0);
}

// ---------------- weight prep ----------------
__global__ void k_zero_h(_Float16* p, int n) {
  int t = blockIdx.x * blockDim.x + threadIdx.x;
  if (t < n) p[t] = (_Float16)0.f;
}

__global__ void k_cvt(const float* __restrict__ src, _Float16* __restrict__ dst,
                      int rows, int cols, int dstStride, int dstOff) {
  int t = blockIdx.x * blockDim.x + threadIdx.x;
  if (t >= rows * cols) return;
  int r = t / cols, c = t - r * cols;
  dst[r * dstStride + dstOff + c] = (_Float16)src[t];
}

// ---------------- stage 1: projection + bilinear fetch + view avg/var ----------------
// block = one (b,n) point, threads = channels. Output x224[b][n][0..223] f16, point-major.
__global__ void k_fetch(const float* __restrict__ feat, const float* __restrict__ wp,
                        const float* __restrict__ intr, const float* __restrict__ extr,
                        _Float16* __restrict__ x224) {
  int bn = blockIdx.x;
  int b = bn / NP, n = bn - b * NP;
  int c = threadIdx.x;
  if (c >= NC) return;
  float X = wp[(long)b * 3 * NP + 0 * NP + n];
  float Y = wp[(long)b * 3 * NP + 1 * NP + n];
  float Z = wp[(long)b * 3 * NP + 2 * NP + n];
  float s = 0.f, ss = 0.f;
  for (int v = 0; v < NV; ++v) {
    const float* E = extr + (long)(b * NV + v) * 12;
    float p0 = E[0] * X + E[1] * Y + E[2]  * Z + E[3];
    float p1 = E[4] * X + E[5] * Y + E[6]  * Z + E[7];
    float p2 = E[8] * X + E[9] * Y + E[10] * Z + E[11];
    const float* Km = intr + (long)(b * NV + v) * 9;
    float u0 = Km[0] * p0 + Km[1] * p1 + Km[2] * p2;
    float u1 = Km[3] * p0 + Km[4] * p1 + Km[5] * p2;
    float u2 = Km[6] * p0 + Km[7] * p1 + Km[8] * p2;
    float z  = u2;
    float zs = fmaxf(z, 1e-6f);
    float u  = u0 / zs, vv = u1 / zs;
    bool valid = (z > 0.001f) && (u >= 0.f) && (u <= (float)(WF - 1)) &&
                 (vv >= 0.f) && (vv <= (float)(HF - 1));
    u  = fminf(fmaxf(u,  0.f), (float)(WF - 1));
    vv = fminf(fmaxf(vv, 0.f), (float)(HF - 1));
    float x0f = floorf(u), y0f = floorf(vv);
    float x1f = fminf(x0f + 1.f, (float)(WF - 1));
    float y1f = fminf(y0f + 1.f, (float)(HF - 1));
    float wx = u - x0f, wy = vv - y0f;
    int xi0 = (int)x0f, yi0 = (int)y0f, xi1 = (int)x1f, yi1 = (int)y1f;
    const float* F = feat + ((long)(b * NV + v) * NC + c) * (HF * WF);
    float f00 = F[yi0 * WF + xi0], f01 = F[yi0 * WF + xi1];
    float f10 = F[yi1 * WF + xi0], f11 = F[yi1 * WF + xi1];
    float val = f00 * (1.f - wx) * (1.f - wy) + f01 * wx * (1.f - wy) +
                f10 * (1.f - wx) * wy + f11 * wx * wy;
    val = valid ? val : 0.f;
    s += val; ss += val * val;
  }
  float avg = s * (1.f / 3.f);
  float var = ss * (1.f / 3.f) - avg * avg;
  _Float16* o = x224 + ((long)b * NP + n) * 224;
  o[c]      = (_Float16)avg;
  o[NC + c] = (_Float16)var;
}

// ---------------- stage 2: normalized points + positional encodings ----------------
// pn point-major f32 [b][n][3]; writes x0 channels 128..159 (24 encodings + 8 zero pad)
__global__ void k_pncoord(const float* __restrict__ wp, const float* __restrict__ mean_,
                          const float* __restrict__ std_, float* __restrict__ pn,
                          _Float16* __restrict__ x0buf) {
  int t = blockIdx.x * blockDim.x + threadIdx.x;
  if (t >= NB * NP) return;
  int b = t / NP, n = t - b * NP;
  _Float16* o = x0buf + ((long)b * NP + n) * 160;
  #pragma unroll
  for (int d = 0; d < 3; ++d) {
    float v = (wp[(long)b * 3 * NP + d * NP + n] - mean_[b * 3 + d]) / std_[b * 3 + d];
    pn[((long)b * NP + n) * 3 + d] = v;
    float sc = v;
    #pragma unroll
    for (int i = 0; i < 8; ++i) { o[128 + i * 3 + d] = (_Float16)sc; sc *= 2.f; }
  }
  #pragma unroll
  for (int r = 152; r < 160; ++r) o[r] = (_Float16)0.f;
}

// ---------------- stage 3: KNN (top-16 smallest squared distances) ----------------
__global__ void k_knn(const float* __restrict__ pn, int* __restrict__ idxb) {
  __shared__ float sx[256], sy[256], sz[256];
  int t = blockIdx.x * 256 + threadIdx.x;
  int b = t / NP, n = t - b * NP;
  const float* p = pn + (long)b * NP * 3;
  float qx = p[n * 3 + 0], qy = p[n * 3 + 1], qz = p[n * 3 + 2];
  float bd[KK]; int bi[KK];
  #pragma unroll
  for (int j = 0; j < KK; ++j) { bd[j] = 3.4e38f; bi[j] = 0; }
  for (int tile = 0; tile < NP; tile += 256) {
    int m = tile + threadIdx.x;
    sx[threadIdx.x] = p[m * 3 + 0];
    sy[threadIdx.x] = p[m * 3 + 1];
    sz[threadIdx.x] = p[m * 3 + 2];
    __syncthreads();
    for (int j = 0; j < 256; ++j) {
      float dx = qx - sx[j], dy = qy - sy[j], dz = qz - sz[j];
      float d = dx * dx + dy * dy + dz * dz;
      if (d < bd[KK - 1]) {
        bd[KK - 1] = d; bi[KK - 1] = tile + j;
        #pragma unroll
        for (int s = KK - 1; s > 0; --s) {
          if (bd[s] < bd[s - 1]) {
            float td = bd[s]; bd[s] = bd[s - 1]; bd[s - 1] = td;
            int   ti = bi[s]; bi[s] = bi[s - 1]; bi[s - 1] = ti;
          }
        }
      }
    }
    __syncthreads();
  }
  #pragma unroll
  for (int j = 0; j < KK; ++j) idxb[((long)b * NP + n) * KK + j] = bi[j];
}

// ---------------- dense WMMA GEMM: out = relu(W(OxI) @ in), point-major ----------------
template<int O, int I>
__global__ void k_gemm_relu(const _Float16* __restrict__ Wt,
                            const _Float16* __restrict__ in,
                            _Float16* __restrict__ out,
                            int inPitch, int outPitch) {
  constexpr int MT = O / 16;
  constexpr int NT = NP / 16;
  int wave = (int)((blockIdx.x * blockDim.x + threadIdx.x) >> 5);
  int lane = threadIdx.x & 31;
  int l16 = lane & 15, hi = lane >> 4;
  if (wave >= NB * MT * NT) return;
  int b = wave / (MT * NT);
  int rem = wave - b * (MT * NT);
  int mt = rem / NT;
  int nt = rem - mt * NT;
  int col = nt * 16 + l16;
  const _Float16* inp = in + (long)b * NP * inPitch + (long)col * inPitch;
  v8f acc = {0.f, 0.f, 0.f, 0.f, 0.f, 0.f, 0.f, 0.f};
  #pragma unroll
  for (int kt = 0; kt < I / 32; ++kt) {
    v16h a, bf;
    #pragma unroll
    for (int h = 0; h < 16; ++h) {
      a[h]  = Wt[(mt * 16 + l16) * I + kt * 32 + kofA(h, hi)];
      bf[h] = inp[kt * 32 + kofB(h, hi)];
    }
    acc = __builtin_amdgcn_wmma_f32_16x16x32_f16(false, a, false, bf,
                                                 (short)0, acc, false, false);
  }
  _Float16* op = out + (long)b * NP * outPitch + (long)col * outPitch;
  #pragma unroll
  for (int r = 0; r < 8; ++r) {
    float v = acc[r];
    v = v > 0.f ? v : 0.f;
    op[mt * 16 + 8 * hi + r] = (_Float16)v;
  }
}

// ---------------- edge conv via WMMA: one wave per point ----------------
// comb(OxK) = relu([Wc|We](Ox(ICEN+IEDGE)) @ [cen(n) broadcast ; edg(idx(n,k))])
// then max (and optionally mean) over the 16 neighbor columns via lane butterflies.
template<int O, int ICEN, int IEDGE, bool HASMEAN>
__global__ void k_edge(const _Float16* __restrict__ Wst,
                       const _Float16* __restrict__ cen, int cenPitch,
                       const _Float16* __restrict__ edg, int edgPitch,
                       const int* __restrict__ nbr,
                       _Float16* __restrict__ maxOut,
                       _Float16* __restrict__ meanOut, int outPitch) {
  constexpr int ITOT = ICEN + IEDGE;
  constexpr int MT = O / 16;
  int wave = (int)((blockIdx.x * blockDim.x + threadIdx.x) >> 5);
  int lane = threadIdx.x & 31;
  int l16 = lane & 15, hi = lane >> 4;
  if (wave >= NB * NP) return;
  int b = wave / NP;
  int n = wave - b * NP;
  int nb = nbr[((long)b * NP + n) * KK + l16];
  const _Float16* cb = cen + (long)b * NP * cenPitch + (long)n  * cenPitch;
  const _Float16* eb = edg + (long)b * NP * edgPitch + (long)nb * edgPitch;
  const v8f vz = {0.f, 0.f, 0.f, 0.f, 0.f, 0.f, 0.f, 0.f};
  v8f acc[MT];
  #pragma unroll
  for (int mt = 0; mt < MT; ++mt) acc[mt] = vz;
  #pragma unroll
  for (int kt = 0; kt < ITOT / 32; ++kt) {
    v16h bf;
    #pragma unroll
    for (int h = 0; h < 16; ++h) {
      int kk = kt * 32 + kofB(h, hi);
      bf[h] = (ICEN > 0 && kk < ICEN) ? cb[kk] : eb[kk - ICEN];
    }
    #pragma unroll
    for (int mt = 0; mt < MT; ++mt) {
      v16h a;
      #pragma unroll
      for (int h = 0; h < 16; ++h)
        a[h] = Wst[(mt * 16 + l16) * ITOT + kt * 32 + kofA(h, hi)];
      acc[mt] = __builtin_amdgcn_wmma_f32_16x16x32_f16(false, a, false, bf,
                                                       (short)0, acc[mt], false, false);
    }
  }
  _Float16* mo = maxOut + (long)b * NP * outPitch + (long)n * outPitch;
  _Float16* eo = nullptr;
  if (HASMEAN) eo = meanOut + (long)b * NP * outPitch + (long)n * outPitch;
  #pragma unroll
  for (int mt = 0; mt < MT; ++mt) {
    #pragma unroll
    for (int r = 0; r < 8; ++r) {
      float v = acc[mt][r];
      v = v > 0.f ? v : 0.f;
      float mx = v, sm = v;
      #pragma unroll
      for (int off = 1; off < 16; off <<= 1) {
        mx = fmaxf(mx, __shfl_xor(mx, off, 32));
        if (HASMEAN) sm += __shfl_xor(sm, off, 32);
      }
      if (l16 == 0) {
        int row = mt * 16 + 8 * hi + r;
        mo[row] = (_Float16)mx;
        if (HASMEAN) eo[row] = (_Float16)(sm * 0.0625f);
      }
    }
  }
}

// ---------------- final 1x16 projection (no relu) ----------------
__global__ void k_out(const _Float16* __restrict__ y3, const float* __restrict__ Wo,
                      float* __restrict__ out) {
  int t = blockIdx.x * blockDim.x + threadIdx.x;
  if (t >= NB * NP) return;
  const _Float16* y = y3 + (long)t * 16;
  float s = 0.f;
  #pragma unroll
  for (int i = 0; i < 16; ++i) s += Wo[i] * (float)y[i];
  out[t] = s;
}

static inline int gridWaves(int waves) { return (waves * 32 + 255) / 256; }
static inline int gridThreads(int threads) { return (threads + 255) / 256; }

extern "C" void kernel_launch(void* const* d_in, const int* in_sizes, int n_in,
                              void* d_out, int out_size, void* d_ws, size_t ws_size,
                              hipStream_t stream) {
  (void)in_sizes; (void)n_in; (void)out_size; (void)ws_size;
  const float* feat   = (const float*)d_in[0];
  const float* wp     = (const float*)d_in[1];
  const float* intr   = (const float*)d_in[2];
  const float* extr   = (const float*)d_in[3];
  const float* mean_  = (const float*)d_in[4];
  const float* std_   = (const float*)d_in[5];
  const float* W_mlp1 = (const float*)d_in[6];
  const float* W_mlp2 = (const float*)d_in[7];
  const float* W_mlp3 = (const float*)d_in[8];
  const float* W_ec1  = (const float*)d_in[9];
  const float* W_ec2c = (const float*)d_in[10];
  const float* W_ec2e = (const float*)d_in[11];
  const float* W_ec3c = (const float*)d_in[12];
  const float* W_ec3e = (const float*)d_in[13];
  const float* W_fm1  = (const float*)d_in[14];
  const float* W_fm2  = (const float*)d_in[15];
  const float* W_fm3  = (const float*)d_in[16];
  const float* W_out  = (const float*)d_in[17];
  float* out = (float*)d_out;

  char* w = (char*)d_ws;
  auto take = [&](size_t nbytes) {
    char* r = w;
    w += (nbytes + 255) & ~(size_t)255;
    return r;
  };
  // f16 weights
  _Float16* Wm1  = (_Float16*)take((size_t)128 * 224 * 2);
  _Float16* Wm2  = (_Float16*)take((size_t)128 * 128 * 2);
  _Float16* Wm3  = (_Float16*)take((size_t)128 * 128 * 2);
  _Float16* We1p = (_Float16*)take((size_t)32 * 160 * 2);  // zero-padded 152->160
  _Float16* Wce2 = (_Float16*)take((size_t)32 * 64 * 2);   // [Wc2 | We2]
  _Float16* Wce3 = (_Float16*)take((size_t)64 * 128 * 2);  // [Wc3 | We3]
  _Float16* Wf1  = (_Float16*)take((size_t)64 * 224 * 2);
  _Float16* Wf2  = (_Float16*)take((size_t)64 * 64 * 2);
  _Float16* Wf3  = (_Float16*)take((size_t)16 * 64 * 2);
  // activations (point-major f16)
  _Float16* x224 = (_Float16*)take((size_t)NB * NP * 224 * 2);
  _Float16* t1   = (_Float16*)take((size_t)NB * NP * 128 * 2);
  _Float16* t2   = (_Float16*)take((size_t)NB * NP * 128 * 2);
  _Float16* x0b  = (_Float16*)take((size_t)NB * NP * 160 * 2); // 128 mlp + 24 coord + 8 pad
  _Float16* cat  = (_Float16*)take((size_t)NB * NP * 224 * 2); // f1|f2max|f2mean|f3max|f3mean
  _Float16* y1   = (_Float16*)take((size_t)NB * NP * 64 * 2);
  _Float16* y2   = (_Float16*)take((size_t)NB * NP * 64 * 2);
  _Float16* y3   = (_Float16*)take((size_t)NB * NP * 16 * 2);
  float* pn      = (float*)take((size_t)NB * NP * 3 * 4);
  int*   idxb    = (int*)take((size_t)NB * NP * KK * 4);

  // --- weight conversion ---
  k_zero_h<<<gridThreads(32 * 160), 256, 0, stream>>>(We1p, 32 * 160);
  k_cvt<<<gridThreads(128 * 224), 256, 0, stream>>>(W_mlp1, Wm1, 128, 224, 224, 0);
  k_cvt<<<gridThreads(128 * 128), 256, 0, stream>>>(W_mlp2, Wm2, 128, 128, 128, 0);
  k_cvt<<<gridThreads(128 * 128), 256, 0, stream>>>(W_mlp3, Wm3, 128, 128, 128, 0);
  k_cvt<<<gridThreads(32 * 152),  256, 0, stream>>>(W_ec1,  We1p, 32, 152, 160, 0);
  k_cvt<<<gridThreads(32 * 32),   256, 0, stream>>>(W_ec2c, Wce2, 32, 32, 64, 0);
  k_cvt<<<gridThreads(32 * 32),   256, 0, stream>>>(W_ec2e, Wce2, 32, 32, 64, 32);
  k_cvt<<<gridThreads(64 * 64),   256, 0, stream>>>(W_ec3c, Wce3, 64, 64, 128, 0);
  k_cvt<<<gridThreads(64 * 64),   256, 0, stream>>>(W_ec3e, Wce3, 64, 64, 128, 64);
  k_cvt<<<gridThreads(64 * 224),  256, 0, stream>>>(W_fm1,  Wf1, 64, 224, 224, 0);
  k_cvt<<<gridThreads(64 * 64),   256, 0, stream>>>(W_fm2,  Wf2, 64, 64, 64, 0);
  k_cvt<<<gridThreads(16 * 64),   256, 0, stream>>>(W_fm3,  Wf3, 16, 64, 64, 0);

  // --- pipeline ---
  k_fetch<<<NB * NP, 128, 0, stream>>>(feat, wp, intr, extr, x224);
  k_pncoord<<<gridThreads(NB * NP), 256, 0, stream>>>(wp, mean_, std_, pn, x0b);

  k_gemm_relu<128, 224><<<gridWaves(NB * 8 * (NP / 16)), 256, 0, stream>>>(Wm1, x224, t1, 224, 128);
  k_gemm_relu<128, 128><<<gridWaves(NB * 8 * (NP / 16)), 256, 0, stream>>>(Wm2, t1, t2, 128, 128);
  k_gemm_relu<128, 128><<<gridWaves(NB * 8 * (NP / 16)), 256, 0, stream>>>(Wm3, t2, x0b, 128, 160);

  k_knn<<<NB * NP / 256, 256, 0, stream>>>(pn, idxb);

  // f1 = relu(Wec1 @ gather(x0)).max_k -> cat[0..31]
  k_edge<32, 0, 160, false><<<gridWaves(NB * NP), 256, 0, stream>>>(
      We1p, x0b, 160, x0b, 160, idxb, cat, (_Float16*)nullptr, 224);
  // f2 = relu(Wc2@f1 + We2@gather(f1)) -> max cat[32..63], mean cat[64..95]
  k_edge<32, 32, 32, true><<<gridWaves(NB * NP), 256, 0, stream>>>(
      Wce2, cat, 224, cat, 224, idxb, cat + 32, cat + 64, 224);
  // f3 = relu(Wc3@f2 + We3@gather(f2)) -> max cat[96..159], mean cat[160..223]
  k_edge<64, 64, 64, true><<<gridWaves(NB * NP), 256, 0, stream>>>(
      Wce3, cat + 32, 224, cat + 32, 224, idxb, cat + 96, cat + 160, 224);

  k_gemm_relu<64, 224><<<gridWaves(NB * 4 * (NP / 16)), 256, 0, stream>>>(Wf1, cat, y1, 224, 64);
  k_gemm_relu<64, 64><<<gridWaves(NB * 4 * (NP / 16)), 256, 0, stream>>>(Wf2, y1, y2, 64, 64);
  k_gemm_relu<16, 64><<<gridWaves(NB * 1 * (NP / 16)), 256, 0, stream>>>(Wf3, y2, y3, 64, 16);

  k_out<<<gridThreads(NB * NP), 256, 0, stream>>>(y3, W_out, out);
}
